// VQ_19756849562144
// MI455X (gfx1250) — matrix operand order repaired and checked by reading, not compiled, not run on hardware
//
#include <hip/hip_runtime.h>

// ---------------------------------------------------------------------------
// VQ nearest-codebook search for MI455X (gfx1250, wave32, WMMA).
//   x:        [B=16, D=512, T=4096] f32
//   codebook: [K=1024, D=512]       f32
//   out:      quantized [B,D,T] f32  ++  indexes [B,T] (written as float)
// Strategy: cross-term GEMM in bf16 WMMA (v_wmma_f32_16x16x32_bf16),
// running per-lane argmax of (2*cross - |c|^2), single end shuffle-reduce,
// coalesced-in-t gather epilogue. Codebook (1MB bf16) stays L2-resident.
// ---------------------------------------------------------------------------

typedef __attribute__((ext_vector_type(16))) __bf16        v16bf;
typedef __attribute__((ext_vector_type(8)))  float         v8f;
typedef __attribute__((ext_vector_type(4)))  unsigned int  u32x4;

#define BB    16
#define DD    512
#define TT    4096
#define KK    1024
#define TWAVE 32
#define ROWH  520   // LDS row pitch in bf16 elems: 512 + 8 pad (1040B, 16B aligned)

__device__ __forceinline__ unsigned short f2bf(float f) {
  // round-to-nearest-even fp32 -> bf16
  unsigned u = __float_as_uint(f);
  u += 0x7FFFu + ((u >> 16) & 1u);
  return (unsigned short)(u >> 16);
}

// ---------------- kernel 1: codebook -> bf16, and |c_k|^2 ------------------
__global__ __launch_bounds__(128) void vq_prep(const float* __restrict__ cb,
                                               unsigned short* __restrict__ cbB,
                                               float* __restrict__ csq) {
  const int k   = blockIdx.x;        // one codebook row per block
  const int tid = threadIdx.x;       // 128 threads * 4 floats = 512
  const float4 f = *(const float4*)(cb + (size_t)k * DD + (size_t)tid * 4);
  unsigned lo = (unsigned)f2bf(f.x) | ((unsigned)f2bf(f.y) << 16);
  unsigned hi = (unsigned)f2bf(f.z) | ((unsigned)f2bf(f.w) << 16);
  unsigned* p = (unsigned*)(cbB + (size_t)k * DD + (size_t)tid * 4);
  p[0] = lo; p[1] = hi;

  float s = f.x * f.x + f.y * f.y + f.z * f.z + f.w * f.w;
  #pragma unroll
  for (int off = 16; off >= 1; off >>= 1) s += __shfl_xor(s, off, 32);
  __shared__ float part[4];
  if ((tid & 31) == 0) part[tid >> 5] = s;
  __syncthreads();
  if (tid == 0) csq[k] = part[0] + part[1] + part[2] + part[3];
}

// ---------------- kernel 2: search + gather --------------------------------
__global__ __launch_bounds__(32) void vq_main(const float* __restrict__ x,
                                              const float* __restrict__ cb,
                                              const unsigned short* __restrict__ cbB,
                                              const float* __restrict__ csq,
                                              float* __restrict__ outq,
                                              float* __restrict__ outi) {
  __shared__ unsigned short lds_x[TWAVE * ROWH]; // x tile, bf16, [t][d]
  __shared__ int idx_lds[TWAVE];

  const int lane = threadIdx.x;      // 0..31
  const int l16  = lane & 15;
  const int lh   = lane >> 4;        // lane-half (WMMA K/M split)

  const int b  = blockIdx.x >> 7;            // T/TWAVE = 128 tiles per batch
  const int t0 = (blockIdx.x & 127) * TWAVE;

  // ---- stage: x[b][d][t0+lane] -> bf16 LDS [t=lane][d], coalesced reads ----
  const float* xb = x + ((size_t)b * DD) * TT + (size_t)t0 + lane;
  for (int d0 = 0; d0 < DD; d0 += 8) {
    u32x4 v;
    #pragma unroll
    for (int j = 0; j < 4; ++j) {
      float f0 = xb[(size_t)(d0 + 2 * j)     * TT];
      float f1 = xb[(size_t)(d0 + 2 * j + 1) * TT];
      v[j] = (unsigned)f2bf(f0) | ((unsigned)f2bf(f1) << 16);
    }
    *(u32x4*)(lds_x + lane * ROWH + d0) = v;   // 16B aligned, conflict-light
  }

  // running best of score = 2*cross - |c|^2, per (t-subtile s, C-row v).
  // Lane L covers the 64 codes with k % 16 == L%16; reduced across lanes at end.
  float bval[2][8];
  int   bidx[2][8];
  #pragma unroll
  for (int s = 0; s < 2; ++s)
    #pragma unroll
    for (int v = 0; v < 8; ++v) { bval[s][v] = -3.0e38f; bidx[s][v] = 0; }

  // B operand base: codebook row k = kt*32 + ks*16 + l16, d-chunk by lane-half
  const unsigned short* bbase = cbB + (size_t)l16 * DD + lh * 16;
  const v8f vzero = {0.f, 0.f, 0.f, 0.f, 0.f, 0.f, 0.f, 0.f};

  for (int kt = 0; kt < KK / 32; ++kt) {       // 32 codes per pass
    v8f acc[2][2];
    acc[0][0] = vzero; acc[0][1] = vzero; acc[1][0] = vzero; acc[1][1] = vzero;

    const unsigned short* b0 = bbase + (size_t)(kt * 32) * DD;
    const unsigned short* b1 = b0 + (size_t)16 * DD;

    #pragma unroll 4
    for (int d0 = 0; d0 < DD; d0 += 32) {
      // B: 32(d) x 16(k) bf16; per lane 16 contiguous elems (2x b128, L2-hot)
      v16bf B0 = *(const v16bf*)(const void*)(b0 + d0);
      v16bf B1 = *(const v16bf*)(const void*)(b1 + d0);
      #pragma unroll
      for (int s = 0; s < 2; ++s) {
        // A: 16(t) x 32(d) bf16 from LDS; lane m=l16 holds K-chunk
        // d0+lh*16..+15 (contiguous, matching B)
        union { u32x4 u[2]; v16bf bv; } A;
        const unsigned short* ap = lds_x + (s * 16 + l16) * ROWH + lh * 16 + d0;
        A.u[0] = *(const u32x4*)(const void*)ap;
        A.u[1] = *(const u32x4*)(const void*)(ap + 8);
        acc[s][0] = __builtin_amdgcn_wmma_f32_16x16x32_bf16(
            false, A.bv, false, B0, (short)0, acc[s][0], false, false);
        acc[s][1] = __builtin_amdgcn_wmma_f32_16x16x32_bf16(
            false, A.bv, false, B1, (short)0, acc[s][1], false, false);
      }
    }

    // epilogue: fold this pass's 32 codes into the running per-lane best
    #pragma unroll
    for (int ks = 0; ks < 2; ++ks) {
      const int myk = kt * 32 + ks * 16 + l16;
      const float c2 = csq[myk];
      #pragma unroll
      for (int s = 0; s < 2; ++s)
        #pragma unroll
        for (int v = 0; v < 8; ++v) {
          float score = 2.0f * acc[s][ks][v] - c2;
          bool better = (score > bval[s][v]);
          bval[s][v] = better ? score : bval[s][v];
          bidx[s][v] = better ? myk  : bidx[s][v];
        }
    }
  }

  // ---- cross-lane argmax over the 16 k-columns in each lane half ----------
  #pragma unroll
  for (int s = 0; s < 2; ++s)
    #pragma unroll
    for (int v = 0; v < 8; ++v) {
      float bv = bval[s][v];
      int   bi = bidx[s][v];
      #pragma unroll
      for (int off = 8; off >= 1; off >>= 1) {
        float ov = __shfl_xor(bv, off, 32);
        int   oi = __shfl_xor(bi, off, 32);
        if (ov > bv || (ov == bv && oi < bi)) { bv = ov; bi = oi; } // tie -> min k
      }
      if (l16 == 0) {                         // lanes 0 and 16
        int tl = s * 16 + v + 8 * lh;         // t within tile
        idx_lds[tl] = bi;
        outi[(size_t)b * TT + t0 + tl] = (float)bi;
      }
    }

  asm volatile("s_wait_dscnt 0" ::: "memory");

  // ---- gather: out[b][d][t0+lane] = codebook[idx][d], coalesced along t ---
  const int myidx = idx_lds[lane];
  const float* crow = cb + (size_t)myidx * DD;          // L2-hot gather
  float* orow = outq + ((size_t)b * DD) * TT + (size_t)t0 + lane;
  for (int d = 0; d < DD; d += 4) {
    float4 c4 = *(const float4*)(crow + d);
    orow[(size_t)(d + 0) * TT] = c4.x;
    orow[(size_t)(d + 1) * TT] = c4.y;
    orow[(size_t)(d + 2) * TT] = c4.z;
    orow[(size_t)(d + 3) * TT] = c4.w;
  }
}

// ---------------------------------------------------------------------------
extern "C" void kernel_launch(void* const* d_in, const int* in_sizes, int n_in,
                              void* d_out, int out_size, void* d_ws, size_t ws_size,
                              hipStream_t stream) {
  const float* x  = (const float*)d_in[0];   // [B, D, T]
  const float* cb = (const float*)d_in[1];   // [K, D]

  unsigned short* cbB = (unsigned short*)d_ws;                 // K*D bf16 = 1 MB
  float* csq = (float*)((char*)d_ws + (size_t)KK * DD * 2);    // K f32 = 4 KB

  float* outq = (float*)d_out;                                 // B*D*T
  float* outi = (float*)d_out + (size_t)BB * DD * TT;          // B*T

  vq_prep<<<KK, 128, 0, stream>>>(cb, cbB, csq);
  vq_main<<<(BB * TT) / TWAVE, TWAVE, 0, stream>>>(x, cb, cbB, csq, outq, outi);
}
